// SPKAL_65283502899266
// MI455X (gfx1250) — matrix operand order repaired and checked by reading, not compiled, not run on hardware
//
#include <hip/hip_runtime.h>

// ---------------------------------------------------------------------------
// KAN layer as one fused GEMM on the CDNA5 matrix pipe.
//   out[b,o] = sum_i silu(x[b,i])*W[o,i] + sum_{i,k} basis(x[b,i],k)*SW[o,i,k]
// Reduction dim R = IN_F*12 (slot 0 = silu, slots 1..11 = basis k=0..10).
// Key optimization: the uniform KAN grid has disjoint consecutive intervals,
// so the 11-wide basis is ONE-HOT: slot = 1 + floor((x+2.2)/0.4), value =
// sin(pi*frac). A-tile expansion costs 1 exp + 1 sin per (b,i), not 11 sins.
// B^T (1024 x 12288 bf16) packed once per call into d_ws.
// B tile staged with global_load_async_to_lds_b128 (ASYNCcnt path).
// Accumulate f32 via v_wmma_f32_16x16x32_bf16.
// ---------------------------------------------------------------------------

typedef __attribute__((ext_vector_type(16))) __bf16 v16bf;
typedef __attribute__((ext_vector_type(8)))  float  v8f;
typedef __attribute__((ext_vector_type(4)))  int    v4i;

union FragBF {
  uint4  u[2];
  v16bf  v;
};

#define IN_F   1024
#define OUT_F  1024
#define BATCH  4096
#define KSLOT  12
#define KTOT   (IN_F * KSLOT)   // 12288
#define BM     128
#define BN     128
#define BK     96                // 8 KAN groups of 12 == 3 WMMA k-chunks of 32
#define LDT    112               // padded LDS row: 224B, 16B-aligned, <=2-way conflicts

#if defined(__gfx1250__) && __has_builtin(__builtin_amdgcn_global_load_async_to_lds_b128)
#define USE_ASYNC_LDS 1
typedef __attribute__((address_space(1))) v4i gv4i;   // global 128-bit chunk
typedef __attribute__((address_space(3))) v4i lv4i;   // LDS 128-bit chunk
#else
#define USE_ASYNC_LDS 0
#endif

__device__ __forceinline__ unsigned short f2bf(float f) {
  unsigned int u = __float_as_uint(f);
  unsigned int r = u + 0x7FFFu + ((u >> 16) & 1u);   // round-to-nearest-even
  return (unsigned short)(r >> 16);
}

__device__ __forceinline__ void wait_async_lds() {
#if USE_ASYNC_LDS
#if __has_builtin(__builtin_amdgcn_s_wait_asynccnt)
  __builtin_amdgcn_s_wait_asynccnt(0);
#else
  asm volatile("s_wait_asynccnt 0x0" ::: "memory");
#endif
#endif
}

// ---------------------------------------------------------------------------
// Pack B^T (OUT_F x KTOT) bf16 into workspace:
//   Bp[o, i*12 + 0]   = bf16(base_weight[o,i])
//   Bp[o, i*12 + 1+k] = bf16(spline_weight[o,i,k] * spline_scaler[o,i])
// ---------------------------------------------------------------------------
__global__ __launch_bounds__(256) void kan_pack_b(
    const float* __restrict__ bw, const float* __restrict__ sw,
    const float* __restrict__ sc, unsigned short* __restrict__ bp) {
  int idx = blockIdx.x * blockDim.x + threadIdx.x;        // 0 .. OUT_F*IN_F-1
  int o = idx >> 10;
  int i = idx & (IN_F - 1);
  float scale = sc[idx];
  const float* swp = sw + (size_t)idx * 11;

  unsigned short vals[12];
  vals[0] = f2bf(bw[idx]);
#pragma unroll
  for (int k = 0; k < 11; k++) vals[k + 1] = f2bf(swp[k] * scale);

  unsigned long long* dst =
      (unsigned long long*)(bp + (size_t)o * KTOT + (size_t)i * KSLOT);
#pragma unroll
  for (int j = 0; j < 3; j++) {
    unsigned long long p =  (unsigned long long)vals[4 * j + 0]
                         | ((unsigned long long)vals[4 * j + 1] << 16)
                         | ((unsigned long long)vals[4 * j + 2] << 32)
                         | ((unsigned long long)vals[4 * j + 3] << 48);
    dst[j] = p;
  }
}

// ---------------------------------------------------------------------------
// Fused GEMM: 128x128 block tile, 8 waves (2x4), each wave 64x32 = 4x2 frags,
// 24 WMMAs per wave per K-step of 96.
// ---------------------------------------------------------------------------
__global__ __launch_bounds__(256) void kan_wmma_gemm(
    const float* __restrict__ X,               // (BATCH, IN_F) f32
    const unsigned short* __restrict__ B,      // (OUT_F, KTOT) bf16
    float* __restrict__ Out) {                 // (BATCH, OUT_F) f32
  __shared__ __align__(16) unsigned short Alds[BM * LDT];  // 28 KB
  __shared__ __align__(16) unsigned short Blds[BN * LDT];  // 28 KB

  const int t    = threadIdx.x;
  const int lane = t & 31;
  const int w    = t >> 5;       // wave 0..7
  const int wm   = w >> 2;       // 0..1  (M)
  const int wn   = w & 3;        // 0..3  (N)
  const int half = lane >> 4;    // lane half for K split
  const int l16  = lane & 15;

  const int m_base = blockIdx.y * BM;
  const int n_base = blockIdx.x * BN;

  v8f acc[4][2];
#pragma unroll
  for (int mf = 0; mf < 4; mf++)
#pragma unroll
    for (int nf = 0; nf < 2; nf++) acc[mf][nf] = (v8f)(0.0f);

  for (int it = 0; it < KTOT / BK; ++it) {
    const int kt     = it * BK;
    const int i_base = it * (BK / KSLOT);   // 8 input features per tile
    __syncthreads();

    // ---- stage B tile: 128 rows x 96 k (async to LDS when available) ----
#pragma unroll
    for (int c = 0; c < 6; c++) {
      int idx  = t + c * 256;          // 0..1535
      int row  = idx / 12;             // 0..127
      int col8 = idx % 12;             // 8 bf16 per chunk
      const unsigned short* src =
          B + (size_t)(n_base + row) * KTOT + (size_t)kt + (size_t)col8 * 8;
      unsigned short* dstp = &Blds[row * LDT + col8 * 8];
#if USE_ASYNC_LDS
      __builtin_amdgcn_global_load_async_to_lds_b128(
          (gv4i*)src, (lv4i*)dstp, 0, 0);
#else
      *(uint4*)dstp = *(const uint4*)src;
#endif
    }

    // ---- prefetch next B tile into L2 (global_prefetch_b8) ----
    if (kt + BK < KTOT) {
      int row = t >> 1;
      __builtin_prefetch(
          B + (size_t)(n_base + row) * KTOT + (size_t)(kt + BK) + (t & 1) * 48,
          0, 3);
    }

    // ---- stage A tile: one-hot basis expansion, 4 (b,i) groups/thread ----
#pragma unroll
    for (int c = 0; c < 4; c++) {
      int g   = t + c * 256;           // group 0..1023
      int row = g >> 3;                // 0..127
      int gi  = g & 7;                 // group within tile
      int i   = i_base + gi;
      float x  = X[(size_t)(m_base + row) * IN_F + i];
      float sil = x / (1.0f + __expf(-x));          // silu (slot 0)
      float u   = (x + 2.2f) * 2.5f;                // (x - a_0)/h
      float fu  = floorf(u);
      int   ki  = (int)fu;                          // interval index
      float s   = __sinf(3.14159265358979f * (u - fu));
      float sv  = (u >= 0.0f && ki < 11) ? s : 0.0f;
      int   kc  = ki < 0 ? 0 : (ki > 10 ? 10 : ki);
      int   p   = kc + 1;                           // slot 1..11
      unsigned long long sb =
          (unsigned long long)f2bf(sv) << ((p & 3) * 16);
      int wsel = p >> 2;                            // which 4-slot word
      unsigned long long w0 =
          (unsigned long long)f2bf(sil) | (wsel == 0 ? sb : 0ull);
      unsigned long long w1 = (wsel == 1) ? sb : 0ull;
      unsigned long long w2 = (wsel == 2) ? sb : 0ull;
      unsigned long long* dst =
          (unsigned long long*)(&Alds[row * LDT + gi * KSLOT]);
      dst[0] = w0;
      dst[1] = w1;
      dst[2] = w2;
    }

    wait_async_lds();
    __syncthreads();

    // ---- compute: 3 k-chunks of 32, 4x2 WMMA frags per wave ----
#pragma unroll
    for (int kk = 0; kk < 3; kk++) {
      const int ks = kk * 32;
      FragBF a[4], b[2];
#pragma unroll
      for (int mf = 0; mf < 4; mf++) {
        int row = wm * 64 + mf * 16 + l16;
        // lanes 0-15: K 0-7 & 16-23 ; lanes 16-31: K 8-15 & 24-31
        const uint4* pa = (const uint4*)(&Alds[row * LDT + ks + half * 8]);
        a[mf].u[0] = pa[0];
        a[mf].u[1] = pa[2];            // +16 elements
      }
#pragma unroll
      for (int nf = 0; nf < 2; nf++) {
        int col = wn * 32 + nf * 16 + l16;
        // lanes 0-15: K 0-7 & 16-23 ; lanes 16-31: K 8-15 & 24-31
        const uint4* pb = (const uint4*)(&Blds[col * LDT + ks + half * 8]);
        b[nf].u[0] = pb[0];
        b[nf].u[1] = pb[2];
      }
#pragma unroll
      for (int mf = 0; mf < 4; mf++)
#pragma unroll
        for (int nf = 0; nf < 2; nf++)
          acc[mf][nf] = __builtin_amdgcn_wmma_f32_16x16x32_bf16(
              false, a[mf].v, false, b[nf].v, (short)0, acc[mf][nf],
              false, false);
    }
  }

  // ---- epilogue: C layout -> VGPR v holds M = half*8 + v, N = l16 ----
#pragma unroll
  for (int mf = 0; mf < 4; mf++) {
#pragma unroll
    for (int nf = 0; nf < 2; nf++) {
      int col   = n_base + wn * 32 + nf * 16 + l16;
      int rbase = m_base + wm * 64 + mf * 16 + half * 8;
#pragma unroll
      for (int v = 0; v < 8; v++) {
        Out[(size_t)(rbase + v) * OUT_F + col] = acc[mf][nf][v];
      }
    }
  }
}

extern "C" void kernel_launch(void* const* d_in, const int* in_sizes, int n_in,
                              void* d_out, int out_size, void* d_ws, size_t ws_size,
                              hipStream_t stream) {
  const float* x  = (const float*)d_in[0];  // (4096, 1024)
  const float* bw = (const float*)d_in[1];  // (1024, 1024)
  const float* sw = (const float*)d_in[2];  // (1024, 1024, 11)
  const float* sc = (const float*)d_in[3];  // (1024, 1024)
  // d_in[4] = grid: uniform, folded into constants.
  float* out = (float*)d_out;               // (4096, 1024) f32
  unsigned short* bp = (unsigned short*)d_ws;  // (1024, 12288) bf16 ~ 24 MB

  (void)in_sizes; (void)n_in; (void)out_size; (void)ws_size;

  // 1) pack [base_w | spline_w*scaler] -> bf16 B^T
  kan_pack_b<<<(OUT_F * IN_F) / 256, 256, 0, stream>>>(bw, sw, sc, bp);

  // 2) fused basis + GEMM
  dim3 grid(OUT_F / BN, BATCH / BM);  // (8, 32)
  kan_wmma_gemm<<<grid, 256, 0, stream>>>(x, bp, out);
}
